// RealtimeHazeGNN_24060406792588
// MI455X (gfx1250) — compile-verified, matrix operand order checked by reading, not run on hardware
//
#include <hip/hip_runtime.h>
#include <hip/hip_bf16.h>
#include <math.h>

typedef __bf16 bf16_t;
typedef __attribute__((ext_vector_type(16))) __bf16 v16bf;
typedef __attribute__((ext_vector_type(8)))  __bf16 v8bf;
typedef __attribute__((ext_vector_type(8)))  float  v8f;

#define NODES 100000
#define EDGES 800000
#define EM    (EDGES + NODES)   // edges + self loops
#define FIN   64
#define HIDN  128
#define HEADS 4
#define CH    32
#define TB    256

__device__ __forceinline__ bf16_t f2bf(float f) { return (bf16_t)f; }

__device__ __forceinline__ void atomicMaxFloat(float* addr, float val) {
  // ordered int trick: positive floats order as ints, negative as reversed uints
  if (val >= 0.0f) atomicMax((int*)addr, __float_as_int(val));
  else             atomicMin((unsigned int*)addr, __float_as_uint(val));
}

// ---------------------------------------------------------------------------
// C[M x NC] = act( A[M x K] @ W[K x NC] + bias )    bf16 WMMA, f32 accum.
// Block = 256 threads = 8 waves. The whole weight matrix is staged ONCE per
// block into LDS, transposed and converted to bf16 (<=32 KB of the 320 KB
// WGP budget), so B fragments are contiguous 16-byte ds_load_b128 reads.
// Each wave owns a 16-row slab and sweeps all NC/16 column tiles, reusing its
// A fragments (kept in VGPRs) across the whole strip -> K/32 * NC/16 WMMAs
// per wave. M must be a multiple of 16 (100000 = 6250*16). Out-of-range waves
// exit uniformly after the barrier, so EXEC is all-1s around every WMMA.
// ---------------------------------------------------------------------------
template <int K, int NC, int RELU>
__global__ __launch_bounds__(TB)
void gemm_tile_wmma(const float* __restrict__ A, const float* __restrict__ W,
                    const float* __restrict__ bias, float* __restrict__ C, int M)
{
  __shared__ bf16_t wt[NC * K];   // transposed: wt[n*K + k]
  __shared__ float  bs[NC];
  const int tid = threadIdx.x;

  // cooperative stage: W (row-major K x NC, f32) -> wt (bf16, [n][k])
  for (int i = tid; i < K * NC; i += TB) {
    int k = i / NC;
    int n = i - k * NC;
    wt[n * K + k] = f2bf(W[i]);
  }
  for (int i = tid; i < NC; i += TB) bs[i] = bias[i];
  __syncthreads();

  const int wave    = tid >> 5;
  const int lane    = tid & 31;
  const int rowTile = blockIdx.x * (TB / 32) + wave;
  if (rowTile * 16 >= M) return;   // wave-uniform exit

  const int mr = lane & 15;        // A row in tile / B,C column in tile
  const int kh = (lane >> 4) << 3; // 0 or 8 : K-half offset

  // preload all A fragments for this wave's 16-row slab
  const float* arow = A + (size_t)(rowTile * 16 + mr) * K;
  v16bf af[K / 32];
#pragma unroll
  for (int kf = 0; kf < K / 32; ++kf) {
    const int k0 = kf * 32;
#pragma unroll
    for (int i = 0; i < 8; ++i) af[kf][i]     = f2bf(arow[k0 + kh + i]);
#pragma unroll
    for (int i = 0; i < 8; ++i) af[kf][8 + i] = f2bf(arow[k0 + 16 + kh + i]);
  }

  const int mbase = rowTile * 16 + ((lane >> 4) << 3);
#pragma unroll
  for (int nt = 0; nt < NC / 16; ++nt) {
    const int ncol = nt * 16 + mr;
    const bf16_t* wcol = &wt[ncol * K];
    v8f acc = {};
#pragma unroll
    for (int kf = 0; kf < K / 32; ++kf) {
      const int k0 = kf * 32;
      v8bf blo = *(const v8bf*)(wcol + k0 + kh);        // 16B ds_load
      v8bf bhi = *(const v8bf*)(wcol + k0 + 16 + kh);   // 16B ds_load
      v16bf bfrag;
#pragma unroll
      for (int i = 0; i < 8; ++i) { bfrag[i] = blo[i]; bfrag[8 + i] = bhi[i]; }
      acc = __builtin_amdgcn_wmma_f32_16x16x32_bf16(false, af[kf], false, bfrag,
                                                    (short)0, acc, false, false);
    }
    const float bv = bs[ncol];
#pragma unroll
    for (int r = 0; r < 8; ++r) {
      float v = acc[r] + bv;
      if (RELU) v = fmaxf(v, 0.0f);
      C[(size_t)(mbase + r) * NC + ncol] = v;
    }
  }
}

// ---------------------------------------------------------------------------
__global__ __launch_bounds__(TB)
void fill_f32(float* __restrict__ p, float v, long n)
{
  long i = (long)blockIdx.x * blockDim.x + threadIdx.x;
  long stride = (long)gridDim.x * blockDim.x;
  for (; i < n; i += stride) p[i] = v;
}

// ---------------------------------------------------------------------------
// one wave per edge: logits[e][h] = att_h . leakyrelu(xl[src]+xr[dst]) ; seg-max
// ---------------------------------------------------------------------------
__global__ __launch_bounds__(TB)
void edge_logits_kernel(const int* __restrict__ src, const int* __restrict__ dst,
                        const float* __restrict__ xl, const float* __restrict__ xr,
                        const float* __restrict__ att,
                        float* __restrict__ logits, float* __restrict__ mx)
{
  int e    = (int)((blockIdx.x * blockDim.x + threadIdx.x) >> 5);
  int lane = threadIdx.x & 31;
  if (e >= EM) return;
  int s, d;
  if (e < EDGES) { s = src[e]; d = dst[e]; } else { s = e - EDGES; d = s; }
  const float* xls = xl + (size_t)s * HIDN;
  const float* xrd = xr + (size_t)d * HIDN;
  float lg[HEADS];
#pragma unroll
  for (int h = 0; h < HEADS; ++h) {
    float v = xls[h * CH + lane] + xrd[h * CH + lane];
    v = (v > 0.0f) ? v : 0.2f * v;           // leaky relu
    float t = v * att[h * CH + lane];
#pragma unroll
    for (int off = 16; off >= 1; off >>= 1) t += __shfl_xor(t, off, 32);
    lg[h] = t;
  }
  if (lane == 0) {
#pragma unroll
    for (int h = 0; h < HEADS; ++h) {
      logits[(size_t)e * HEADS + h] = lg[h];
      atomicMaxFloat(&mx[(size_t)d * HEADS + h], lg[h]);
    }
  }
}

// one thread per (edge, head): p = exp(logit - max[dst]); seg-sum
__global__ __launch_bounds__(TB)
void edge_exp_kernel(const int* __restrict__ dst, const float* __restrict__ logits,
                     const float* __restrict__ mx, float* __restrict__ p,
                     float* __restrict__ sm)
{
  long idx = (long)blockIdx.x * blockDim.x + threadIdx.x;
  if (idx >= (long)EM * HEADS) return;
  int e = (int)(idx >> 2);
  int h = (int)(idx & 3);
  int d = (e < EDGES) ? dst[e] : (e - EDGES);
  float pe = __expf(logits[idx] - mx[(size_t)d * HEADS + h]);
  p[idx] = pe;
  atomicAdd(&sm[(size_t)d * HEADS + h], pe);
}

// one wave per edge: agg[dst] += alpha * xl[src]
__global__ __launch_bounds__(TB)
void edge_agg_kernel(const int* __restrict__ src, const int* __restrict__ dst,
                     const float* __restrict__ p, const float* __restrict__ sm,
                     const float* __restrict__ xl, float* __restrict__ agg)
{
  int e    = (int)((blockIdx.x * blockDim.x + threadIdx.x) >> 5);
  int lane = threadIdx.x & 31;
  if (e >= EM) return;
  int s, d;
  if (e < EDGES) { s = src[e]; d = dst[e]; } else { s = e - EDGES; d = s; }
  const float* xls = xl + (size_t)s * HIDN;
  float* aggd = agg + (size_t)d * HIDN;
#pragma unroll
  for (int h = 0; h < HEADS; ++h) {
    float alpha = p[(size_t)e * HEADS + h] / sm[(size_t)d * HEADS + h];
    atomicAdd(&aggd[h * CH + lane], alpha * xls[h * CH + lane]);
  }
}

// ---------------------------------------------------------------------------
// one wave per node: out = elu( [hprev +] LN(agg + bias) * g + b )
// ---------------------------------------------------------------------------
__global__ __launch_bounds__(TB)
void ln_act_kernel(const float* __restrict__ agg, const float* __restrict__ bias,
                   const float* __restrict__ gamma, const float* __restrict__ beta,
                   const float* __restrict__ hprev, float* __restrict__ out)
{
  int i    = (int)((blockIdx.x * blockDim.x + threadIdx.x) >> 5);
  int lane = threadIdx.x & 31;
  if (i >= NODES) return;
  const float* row = agg + (size_t)i * HIDN;
  float v[4];
  float s = 0.0f;
#pragma unroll
  for (int j = 0; j < 4; ++j) {
    int c = lane + 32 * j;
    v[j] = row[c] + bias[c];
    s += v[j];
  }
#pragma unroll
  for (int off = 16; off >= 1; off >>= 1) s += __shfl_xor(s, off, 32);
  float mu = s * (1.0f / 128.0f);
  float q = 0.0f;
#pragma unroll
  for (int j = 0; j < 4; ++j) { float dlt = v[j] - mu; q += dlt * dlt; }
#pragma unroll
  for (int off = 16; off >= 1; off >>= 1) q += __shfl_xor(q, off, 32);
  float rstd = rsqrtf(q * (1.0f / 128.0f) + 1e-5f);
#pragma unroll
  for (int j = 0; j < 4; ++j) {
    int c = lane + 32 * j;
    float y = (v[j] - mu) * rstd * gamma[c] + beta[c];
    if (hprev) y += hprev[(size_t)i * HIDN + c];
    out[(size_t)i * HIDN + c] = (y > 0.0f) ? y : (__expf(y) - 1.0f);  // elu
  }
}

// ---------------------------------------------------------------------------
// one wave per node: pred = hp1 . Wp2 + bp2 ; unc = softplus(hu1 . Wu2 + bu2)
// ---------------------------------------------------------------------------
__global__ __launch_bounds__(TB)
void heads_kernel(const float* __restrict__ hp1, const float* __restrict__ Wp2,
                  const float* __restrict__ bp2, const float* __restrict__ hu1,
                  const float* __restrict__ Wu2, const float* __restrict__ bu2,
                  float* __restrict__ out)
{
  int i    = (int)((blockIdx.x * blockDim.x + threadIdx.x) >> 5);
  int lane = threadIdx.x & 31;
  if (i >= NODES) return;
  float sp = hp1[(size_t)i * 64 + lane]      * Wp2[lane] +
             hp1[(size_t)i * 64 + 32 + lane] * Wp2[32 + lane];
  float su = hu1[(size_t)i * 32 + lane]      * Wu2[lane];
#pragma unroll
  for (int off = 16; off >= 1; off >>= 1) {
    sp += __shfl_xor(sp, off, 32);
    su += __shfl_xor(su, off, 32);
  }
  if (lane == 0) {
    out[i] = sp + bp2[0];
    float xv = su + bu2[0];
    out[NODES + i] = (xv > 20.0f) ? xv : log1pf(__expf(xv));  // softplus
  }
}

// ---------------------------------------------------------------------------
extern "C" void kernel_launch(void* const* d_in, const int* in_sizes, int n_in,
                              void* d_out, int out_size, void* d_ws, size_t ws_size,
                              hipStream_t stream)
{
  const float* x    = (const float*)d_in[0];
  const int*   ei   = (const int*)d_in[1];
  const int*   src  = ei;
  const int*   dst  = ei + EDGES;
  const float* Wl1  = (const float*)d_in[2];
  const float* bl1  = (const float*)d_in[3];
  const float* Wr1  = (const float*)d_in[4];
  const float* br1  = (const float*)d_in[5];
  const float* att1 = (const float*)d_in[6];
  const float* bias1= (const float*)d_in[7];
  const float* g1   = (const float*)d_in[8];
  const float* be1  = (const float*)d_in[9];
  const float* Wl2  = (const float*)d_in[10];
  const float* bl2  = (const float*)d_in[11];
  const float* Wr2  = (const float*)d_in[12];
  const float* br2  = (const float*)d_in[13];
  const float* att2 = (const float*)d_in[14];
  const float* bias2= (const float*)d_in[15];
  const float* g2   = (const float*)d_in[16];
  const float* be2  = (const float*)d_in[17];
  const float* Wp1  = (const float*)d_in[18];
  const float* bp1  = (const float*)d_in[19];
  const float* Wp2  = (const float*)d_in[20];
  const float* bp2  = (const float*)d_in[21];
  const float* Wu1  = (const float*)d_in[22];
  const float* bu1  = (const float*)d_in[23];
  const float* Wu2  = (const float*)d_in[24];
  const float* bu2  = (const float*)d_in[25];
  float* out = (float*)d_out;

  // workspace carve (floats)
  float* ws = (float*)d_ws;
  size_t o = 0;
  float* bufA = ws + o; o += (size_t)NODES * HIDN;   // xl / hp1
  float* bufB = ws + o; o += (size_t)NODES * HIDN;   // xr / hu1
  float* bufC = ws + o; o += (size_t)NODES * HIDN;   // aggregate / final feats
  float* bufD = ws + o; o += (size_t)NODES * HIDN;   // h after layer 1
  float* pbuf = ws + o; o += (size_t)EM * HEADS;     // per-edge probs
  float* mx   = ws + o; o += (size_t)NODES * HEADS;  // segment max
  float* sm   = ws + o; o += (size_t)NODES * HEADS;  // segment sum

  const int gemmBlocks = (NODES / 16 + (TB / 32) - 1) / (TB / 32);  // 8 row-tiles/block
  const int edgeBlocks = (EM * 32 + TB - 1) / TB;
  const int expBlocks  = (int)(((long)EM * HEADS + TB - 1) / TB);
  const int nodeBlocks = (NODES * 32 + TB - 1) / TB;

  // ---- GATv2 layer 1 ----
  gemm_tile_wmma<FIN, HIDN, 0><<<gemmBlocks, TB, 0, stream>>>(x, Wl1, bl1, bufA, NODES);
  gemm_tile_wmma<FIN, HIDN, 0><<<gemmBlocks, TB, 0, stream>>>(x, Wr1, br1, bufB, NODES);
  fill_f32<<<1024, TB, 0, stream>>>(bufC, 0.0f, (long)NODES * HIDN);
  fill_f32<<<256,  TB, 0, stream>>>(mx, -INFINITY, (long)NODES * HEADS);
  fill_f32<<<256,  TB, 0, stream>>>(sm, 0.0f, (long)NODES * HEADS);
  edge_logits_kernel<<<edgeBlocks, TB, 0, stream>>>(src, dst, bufA, bufB, att1, pbuf, mx);
  edge_exp_kernel<<<expBlocks, TB, 0, stream>>>(dst, pbuf, mx, pbuf, sm);
  edge_agg_kernel<<<edgeBlocks, TB, 0, stream>>>(src, dst, pbuf, sm, bufA, bufC);
  ln_act_kernel<<<nodeBlocks, TB, 0, stream>>>(bufC, bias1, g1, be1, nullptr, bufD);

  // ---- GATv2 layer 2 ----
  gemm_tile_wmma<HIDN, HIDN, 0><<<gemmBlocks, TB, 0, stream>>>(bufD, Wl2, bl2, bufA, NODES);
  gemm_tile_wmma<HIDN, HIDN, 0><<<gemmBlocks, TB, 0, stream>>>(bufD, Wr2, br2, bufB, NODES);
  fill_f32<<<1024, TB, 0, stream>>>(bufC, 0.0f, (long)NODES * HIDN);
  fill_f32<<<256,  TB, 0, stream>>>(mx, -INFINITY, (long)NODES * HEADS);
  fill_f32<<<256,  TB, 0, stream>>>(sm, 0.0f, (long)NODES * HEADS);
  edge_logits_kernel<<<edgeBlocks, TB, 0, stream>>>(src, dst, bufA, bufB, att2, pbuf, mx);
  edge_exp_kernel<<<expBlocks, TB, 0, stream>>>(dst, pbuf, mx, pbuf, sm);
  edge_agg_kernel<<<edgeBlocks, TB, 0, stream>>>(src, dst, pbuf, sm, bufA, bufC);
  // hf = elu(h + LN(agg + bias2))  (in place over bufC, reads own row only)
  ln_act_kernel<<<nodeBlocks, TB, 0, stream>>>(bufC, bias2, g2, be2, bufD, bufC);

  // ---- heads ----
  gemm_tile_wmma<HIDN, 64, 1><<<gemmBlocks, TB, 0, stream>>>(bufC, Wp1, bp1, bufA, NODES);
  gemm_tile_wmma<HIDN, 32, 1><<<gemmBlocks, TB, 0, stream>>>(bufC, Wu1, bu1, bufB, NODES);
  heads_kernel<<<nodeBlocks, TB, 0, stream>>>(bufA, Wp2, bp2, bufB, Wu2, bu2, out);

  (void)in_sizes; (void)n_in; (void)out_size; (void)ws_size;
}
